// RayMarchingSampler_43087111914122
// MI455X (gfx1250) — matrix operand order repaired and checked by reading, not compiled, not run on hardware
//
#include <hip/hip_runtime.h>
#include <math.h>

// ---- problem constants (match reference) ----
#define N_RAYS    32768
#define N_STEPS   128
#define STEP      0.01f
#define RES       128
#define AABB_MIN  (-2.0f)
#define AABB_SIZE 4.0f
#define SIGMA_EPS 0.01f

typedef _Float16 h16 __attribute__((ext_vector_type(16)));
typedef _Float16 h8  __attribute__((ext_vector_type(8)));
typedef float    f8  __attribute__((ext_vector_type(8)));

#define WAVES_PER_BLOCK 8   // 256 threads, wave32

// Wave-local LDS store->load hand-off: LDS ops from one wave execute in order
// (CDNA5 ISA ch.11); wave_barrier stops compiler reordering, s_wait_dscnt 0
// guarantees the stores have landed in LDS before dependent loads issue.
__device__ __forceinline__ void wave_lds_fence() {
    __builtin_amdgcn_wave_barrier();
    asm volatile("s_wait_dscnt 0x0" ::: "memory");
    __builtin_amdgcn_wave_barrier();
}

__global__ __launch_bounds__(256)
void RayMarchingSampler_kernel(const float* __restrict__ rays,
                               const float* __restrict__ occ_grid,
                               const float* __restrict__ jitter,
                               const float* __restrict__ W1,
                               const float* __restrict__ b1,
                               const float* __restrict__ W2,
                               const float* __restrict__ b2,
                               float* __restrict__ out)
{
    // per-wave scratch: relu(h) tile (16 samples x 32 hidden, f16) + sigma bcast
    __shared__ _Float16 hbuf[WAVES_PER_BLOCK][16 * 32];
    __shared__ float    sbuf[WAVES_PER_BLOCK][16];

    const int lane = threadIdx.x & 31;
    const int wave = threadIdx.x >> 5;
    const int ray  = blockIdx.x * WAVES_PER_BLOCK + wave;

    const int  nn  = lane & 15;        // sample-in-tile / matrix column index
    const bool hiK = lane >= 16;       // upper lane half covers shifted K ranges

    // ---- per-ray scalars (uniform across the wave) ----
    const float* rp = rays + (size_t)ray * 8;
    const float ox = rp[0], oy = rp[1], oz = rp[2];
    const float dx = rp[3], dy = rp[4], dz = rp[5];
    const float farv = rp[7];
    const float t0   = rp[6] + jitter[ray] * STEP;

    // ---- B matrices (constant across tiles) ----
    // 16-bit B 32x16 layout: lane n in 0..15 holds col n, elems 0..7 = K 0..7, 8..15 = K 16..23;
    // lanes 16..31 hold K 8..15 / 24..31. Layer-1 rows: K0..2 = W1, K3 = b1 (A puts 1.0
    // at K=3), K>=4 zero.
    h16 bW1lo = {}; h16 bW1hi = {}; h16 bW2 = {};
    if (!hiK) {
        bW1lo[0] = (_Float16)W1[0 * 32 + nn];
        bW1lo[1] = (_Float16)W1[1 * 32 + nn];
        bW1lo[2] = (_Float16)W1[2 * 32 + nn];
        bW1lo[3] = (_Float16)b1[nn];
        bW1hi[0] = (_Float16)W1[0 * 32 + 16 + nn];
        bW1hi[1] = (_Float16)W1[1 * 32 + 16 + nn];
        bW1hi[2] = (_Float16)W1[2 * 32 + 16 + nn];
        bW1hi[3] = (_Float16)b1[16 + nn];
    }
    // Layer 2: W2 in column 0 only (other columns stay zero -> D cols>0 zero).
    if (nn == 0) {
        #pragma unroll
        for (int e = 0; e < 16; ++e)
            bW2[e] = (_Float16)W2[(e < 8 ? 0 : 8) + (hiK ? 8 : 0) + e];
    }
    const float bias2 = b2[0];

    const f8 zeroC = {};

    #pragma unroll 1
    for (int tile = 0; tile < N_STEPS / 16; ++tile) {
        const int s0 = tile * 16;

        // ---- lanes 0..15: geometry for sample m = nn ----
        const float tstart = t0 + (float)(s0 + nn) * STEP;
        const float tmid   = tstart + 0.5f * STEP;
        const float tend   = tstart + STEP;
        const float x = ox + dx * tmid;
        const float y = oy + dy * tmid;
        const float z = oz + dz * tmid;

        const float ux = (x - AABB_MIN) / AABB_SIZE;
        const float uy = (y - AABB_MIN) / AABB_SIZE;
        const float uz = (z - AABB_MIN) / AABB_SIZE;
        const bool in_aabb = (ux >= 0.f) & (ux < 1.f) &
                             (uy >= 0.f) & (uy < 1.f) &
                             (uz >= 0.f) & (uz < 1.f);
        const int i0 = min(max((int)floorf(ux * (float)RES), 0), RES - 1);
        const int i1 = min(max((int)floorf(uy * (float)RES), 0), RES - 1);
        const int i2 = min(max((int)floorf(uz * (float)RES), 0), RES - 1);
        const float occv = occ_grid[((size_t)i0 * RES + i1) * RES + i2]; // L2-resident
        const bool occb = occv > 0.5f;

        // ---- Layer 1 WMMA: h = relu(xyz @ W1 + b1) ----
        // 16-bit A 16x32 layout: lanes 0..15 hold M=lane, K=0..7 in elems 0..7
        // (x,y,z,1 at K=0..3); lanes 16..31 cover K=8..15/24..31 -> zero.
        h16 a1 = {};
        if (!hiK) {
            a1[0] = (_Float16)x;
            a1[1] = (_Float16)y;
            a1[2] = (_Float16)z;
            a1[3] = (_Float16)1.0f;
        }
        f8 c0 = __builtin_amdgcn_wmma_f32_16x16x32_f16(
            false, a1, false, bW1lo, (short)0, zeroC, false, false);
        f8 c1 = __builtin_amdgcn_wmma_f32_16x16x32_f16(
            false, a1, false, bW1hi, (short)0, zeroC, false, false);

        // ---- transpose relu(h): C-layout -> LDS [m][n] (f16) ----
        // C/D layout: lane l holds col N=l%16, VGPR r holds row M = r + 8*(l/16).
        _Float16* hb = hbuf[wave];
        {
            const int mbase = hiK ? 8 : 0;
            #pragma unroll
            for (int r = 0; r < 8; ++r) {
                const int m = mbase + r;
                const float r0 = c0[r] > 0.f ? c0[r] : 0.f;
                const float r1 = c1[r] > 0.f ? c1[r] : 0.f;
                hb[m * 32 + nn]      = (_Float16)r0;
                hb[m * 32 + 16 + nn] = (_Float16)r1;
            }
        }
        wave_lds_fence();

        // A2 layout: lane l holds M=l%16; elems 0..7 = K(0..7 | 8..15),
        // elems 8..15 = K(16..23 | 24..31) by lane half. 16B-aligned b128 loads.
        const int koff = hiK ? 8 : 0;
        const h8 alo = *(const h8*)(hb + nn * 32 + koff);
        const h8 ahi = *(const h8*)(hb + nn * 32 + 16 + koff);
        h16 a2;
        #pragma unroll
        for (int e = 0; e < 8; ++e) { a2[e] = alo[e]; a2[8 + e] = ahi[e]; }

        // ---- Layer 2 WMMA: sigma_pre = h @ W2 ----
        f8 d2 = __builtin_amdgcn_wmma_f32_16x16x32_f16(
            false, a2, false, bW2, (short)0, zeroC, false, false);

        // broadcast sigma (lives in lanes 0 and 16; VGPR r = sample r / 8+r)
        float* sb = sbuf[wave];
        if (nn == 0) {
            const int mbase = hiK ? 8 : 0;
            #pragma unroll
            for (int r = 0; r < 8; ++r) sb[mbase + r] = d2[r];
        }
        wave_lds_fence();

        // ---- mask + masked xyz, streamed out with NT stores (no L2 pollution,
        // keep the occ grid resident) ----
        if (!hiK) {
            const float xp = sb[nn] + bias2;
            const float sigma = (xp > 20.f) ? xp : log1pf(expf(xp)); // softplus
            const bool m = (tend <= farv) & in_aabb & occb & (sigma > SIGMA_EPS);
            const float w = m ? 1.f : 0.f;
            float* op = out + ((size_t)ray * N_STEPS + s0 + nn) * 3;
            __builtin_nontemporal_store(x * w, op + 0);
            __builtin_nontemporal_store(y * w, op + 1);
            __builtin_nontemporal_store(z * w, op + 2);
        }
        wave_lds_fence();   // WAR guard: next tile's hbuf/sbuf writes
    }
}

extern "C" void kernel_launch(void* const* d_in, const int* in_sizes, int n_in,
                              void* d_out, int out_size, void* d_ws, size_t ws_size,
                              hipStream_t stream) {
    (void)in_sizes; (void)n_in; (void)d_ws; (void)ws_size; (void)out_size;
    const float* rays     = (const float*)d_in[0];
    const float* occ_grid = (const float*)d_in[1];
    const float* jitter   = (const float*)d_in[2];
    const float* W1       = (const float*)d_in[3];
    const float* b1       = (const float*)d_in[4];
    const float* W2       = (const float*)d_in[5];
    const float* b2       = (const float*)d_in[6];
    float* out = (float*)d_out;

    dim3 grid(N_RAYS / WAVES_PER_BLOCK);   // 4096 blocks x 8 waves = 32768 rays
    dim3 block(32 * WAVES_PER_BLOCK);
    RayMarchingSampler_kernel<<<grid, block, 0, stream>>>(
        rays, occ_grid, jitter, W1, b1, W2, b2, out);
}
